// SwinTransformer_22462678958667
// MI455X (gfx1250) — matrix operand order repaired and checked by reading, not compiled, not run on hardware
//
#include <hip/hip_runtime.h>

typedef _Float16 half_t;
typedef __attribute__((ext_vector_type(16))) _Float16 v16h;
typedef __attribute__((ext_vector_type(8)))  _Float16 v8h;
typedef __attribute__((ext_vector_type(8)))  float    v8f;

#define HW        56
#define CDIM      128
#define HEADS     4
#define DHEAD     32
#define WIN       7
#define SHIFTV    3
#define LPAD      64
#define XS_STRIDE 136   // halves; 272B rows (16B aligned)
#define VT_STRIDE 72    // halves; 144B rows (16B aligned)
#define AT_STRIDE 72

static __device__ __forceinline__ v8f wmma_f16(v16h a, v16h b, v8f c) {
  return __builtin_amdgcn_wmma_f32_16x16x32_f16(false, a, false, b, (short)0, c, false, false);
}

// A-matrix fragment: storage [M,K] row-major.
// lanes 0-15: M=lane, K = {cb+0..7, cb+16..23}; lanes 16-31: M=lane-16, K = {cb+8..15, cb+24..31}
template <typename P>
static __device__ __forceinline__ v16h frag_A(P p, int stride, int row_base, int col_base) {
  const int lane = threadIdx.x & 31;
  const int hi = lane >> 4;
  const half_t* r = p + (row_base + (lane & 15)) * stride + col_base + hi * 8;
  union { v16h v; v8h h[2]; } u;
  u.h[0] = *(const v8h*)(r);
  u.h[1] = *(const v8h*)(r + 16);
  return u.v;
}

// B-matrix fragment: storage [N,K] row-major (= weight [out,in]).
// lanes 0-15: N=lane, K = {kb+0..7, kb+16..23}; lanes 16-31: N=lane-16, K = {kb+8..15, kb+24..31}
template <typename P>
static __device__ __forceinline__ v16h frag_B(P p, int stride, int n_base, int k_base) {
  const int lane = threadIdx.x & 31;
  const int hi = lane >> 4;
  const half_t* r = p + (n_base + (lane & 15)) * stride + k_base + hi * 8;
  union { v16h v; v8h h[2]; } u;
  u.h[0] = *(const v8h*)(r);
  u.h[1] = *(const v8h*)(r + 16);
  return u.v;
}

// region label for the cyclic-shift mask (rows/cols of the rolled image)
static __device__ __forceinline__ int lab(int v) { return v < HW - WIN ? 0 : (v < HW - SHIFTV ? 1 : 2); }

__global__ void swin_convert_w(const float* __restrict__ qkvw, const float* __restrict__ projw,
                               half_t* __restrict__ qkvw_h, half_t* __restrict__ projw_h) {
  int i = blockIdx.x * 256 + threadIdx.x;
  if (i < 3 * CDIM * CDIM) qkvw_h[i] = (half_t)qkvw[i];
  if (i < CDIM * CDIM)     projw_h[i] = (half_t)projw[i];
}

__global__ __launch_bounds__(256) void swin_window_attn(
    const float* __restrict__ x,
    const float* __restrict__ qkv_bias,
    const float* __restrict__ proj_bias,
    const float* __restrict__ rel_bias,      // [169][4]
    const half_t* __restrict__ qkvw_h,       // [384][128]  = [N,K]
    const half_t* __restrict__ projw_h,      // [128][128]  = [N,K]
    float* __restrict__ out) {
  extern __shared__ __align__(16) char smem_raw[];
  half_t* XS   = (half_t*)smem_raw;                 // [64][136] window input (f16), reused for attn-out
  half_t* Qs   = XS + LPAD * XS_STRIDE;             // [64][136] q (pre-scaled)
  half_t* Ks   = Qs + LPAD * XS_STRIDE;             // [64][136] k
  half_t* VT   = Ks + LPAD * XS_STRIDE;             // [128][72] v transposed [d_total, Lpad]
  half_t* ATTN = VT + CDIM * VT_STRIDE;             // [4][64][72] softmax(attn) per head

  const int wid  = blockIdx.x;
  const int b    = wid >> 6;
  const int win  = wid & 63;
  const int wy   = win >> 3;
  const int wx   = win & 7;
  const int lane = threadIdx.x & 31;
  const int wave = threadIdx.x >> 5;
  const int hi   = lane >> 4;
  const int nl   = lane & 15;
  const int mt   = wave & 3;          // fixed M-tile per wave for the dense GEMM phases
  const int nh   = wave >> 2;         // which N-half this wave covers

  // warm L2/WGP$ with the converted weights (global_prefetch_b8)
  __builtin_prefetch(qkvw_h + (size_t)threadIdx.x * 192, 0, 1);
  __builtin_prefetch(projw_h + (size_t)threadIdx.x * 64, 0, 1);

  // ---- Phase 1: cyclic-shift gather + window partition, f32 -> f16 into LDS ----
  for (int i = threadIdx.x; i < LPAD * 32; i += 256) {
    int row = i >> 5;
    int cg  = (i & 31) << 2;
    half_t* dst = XS + row * XS_STRIDE + cg;
    if (row < 49) {
      int sy = (wy * WIN + row / 7 + SHIFTV) % HW;   // roll(x, -s)
      int sx = (wx * WIN + row % 7 + SHIFTV) % HW;
      float4 v = *(const float4*)(x + (((size_t)b * HW + sy) * HW + sx) * CDIM + cg);
      dst[0] = (half_t)v.x; dst[1] = (half_t)v.y; dst[2] = (half_t)v.z; dst[3] = (half_t)v.w;
    } else {
      dst[0] = (half_t)0.f; dst[1] = (half_t)0.f; dst[2] = (half_t)0.f; dst[3] = (half_t)0.f;
    }
  }
  __syncthreads();

  // ---- Phase 2: QKV GEMM  [64,128] x W^T -> q/k natural, v transposed ----
  {
    v16h a[4];
#pragma unroll
    for (int ks = 0; ks < 4; ++ks) a[ks] = frag_A(XS, XS_STRIDE, mt * 16, ks * 32);  // hoisted per wave
    for (int j = 0; j < 12; ++j) {
      int nt = nh * 12 + j;            // 0..23, each (mt,nt) covered exactly once
      v8f acc = {0, 0, 0, 0, 0, 0, 0, 0};
#pragma unroll
      for (int ks = 0; ks < 4; ++ks) {
        v16h bm = frag_B(qkvw_h, CDIM, nt * 16, ks * 32);
        acc = wmma_f16(a[ks], bm, acc);
      }
      float bias = qkv_bias[nt * 16 + nl];
      if (nt < 8) {                    // q: scale by d^-0.5, store natural
        const float scale = 0.17677669529663687f;
        half_t* base = Qs + (mt * 16 + hi * 8) * XS_STRIDE + nt * 16 + nl;
#pragma unroll
        for (int r = 0; r < 8; ++r) base[r * XS_STRIDE] = (half_t)((acc[r] + bias) * scale);
      } else if (nt < 16) {            // k: store natural ([L,d] == [N,K] for q.k^T)
        half_t* base = Ks + (mt * 16 + hi * 8) * XS_STRIDE + (nt - 8) * 16 + nl;
#pragma unroll
        for (int r = 0; r < 8; ++r) base[r * XS_STRIDE] = (half_t)(acc[r] + bias);
      } else {                         // v: transposed packed store -> VT[d, Lpad]
        v8h h;
#pragma unroll
        for (int r = 0; r < 8; ++r) h[r] = (half_t)(acc[r] + bias);
        *(v8h*)(VT + ((nt - 16) * 16 + nl) * VT_STRIDE + mt * 16 + hi * 8) = h;
      }
    }
  }
  __syncthreads();

  // ---- Phase 3: attn = softmax(q k^T + rel_bias + shift_mask), in-register softmax ----
  for (int j = 0; j < 2; ++j) {
    int job  = wave + (j << 3);       // 16 jobs: (head, Mtile)
    int head = job >> 2;
    int amt  = job & 3;
    v8f t[4];
    v16h a = frag_A(Qs, XS_STRIDE, amt * 16, head * DHEAD);
#pragma unroll
    for (int ntt = 0; ntt < 4; ++ntt) {
      v16h bm = frag_B(Ks, XS_STRIDE, ntt * 16, head * DHEAD);
      v8f c = {0, 0, 0, 0, 0, 0, 0, 0};
      t[ntt] = wmma_f16(a, bm, c);    // K = d = 32 in one WMMA
    }
#pragma unroll
    for (int ntt = 0; ntt < 4; ++ntt) {
      int N = ntt * 16 + nl;
      int ky = N / 7, kx = N % 7;
      int lkr = lab(wy * WIN + ky), lkc = lab(wx * WIN + kx);
#pragma unroll
      for (int r = 0; r < 8; ++r) {
        int M = amt * 16 + hi * 8 + r;
        float av = t[ntt][r];
        if (N >= 49) {
          av = -1e30f;                 // key padding -> exp()=0
        } else if (M < 49) {
          int qy = M / 7, qx = M % 7;
          av += rel_bias[((qy - ky + 6) * 13 + (qx - kx + 6)) * HEADS + head];
          int lq = lab(wy * WIN + qy) * 3 + lab(wx * WIN + qx);
          if (lq != lkr * 3 + lkc) av -= 100.f;
        }
        t[ntt][r] = av;
      }
    }
#pragma unroll
    for (int r = 0; r < 8; ++r) {      // row softmax: reduce across 16-lane half-wave
      float m = fmaxf(fmaxf(t[0][r], t[1][r]), fmaxf(t[2][r], t[3][r]));
#pragma unroll
      for (int s = 1; s < 16; s <<= 1) m = fmaxf(m, __shfl_xor(m, s, 32));
      float sum = 0.f;
#pragma unroll
      for (int ntt = 0; ntt < 4; ++ntt) { float e = __expf(t[ntt][r] - m); t[ntt][r] = e; sum += e; }
#pragma unroll
      for (int s = 1; s < 16; s <<= 1) sum += __shfl_xor(sum, s, 32);
      float inv = 1.f / sum;
#pragma unroll
      for (int ntt = 0; ntt < 4; ++ntt) t[ntt][r] *= inv;
    }
    half_t* base = ATTN + head * (LPAD * AT_STRIDE) + (amt * 16 + hi * 8) * AT_STRIDE;
#pragma unroll
    for (int ntt = 0; ntt < 4; ++ntt)
#pragma unroll
      for (int r = 0; r < 8; ++r)
        base[r * AT_STRIDE + ntt * 16 + nl] = (half_t)t[ntt][r];
  }
  __syncthreads();

  // ---- Phase 4: out = attn @ v  (A = attn [L,Lpad], B = VT [d, Lpad]) -> reuse XS ----
  for (int i = 0; i < 4; ++i) {
    int combo = nh * 4 + i;           // (head, local n-tile); mt fixed per wave
    int head  = combo >> 1;
    int ntl   = combo & 1;
    const half_t* Ah = ATTN + head * (LPAD * AT_STRIDE);
    v8f acc = {0, 0, 0, 0, 0, 0, 0, 0};
#pragma unroll
    for (int ks = 0; ks < 2; ++ks) {
      v16h a  = frag_A(Ah, AT_STRIDE, mt * 16, ks * 32);
      v16h bm = frag_B(VT, VT_STRIDE, head * DHEAD + ntl * 16, ks * 32);
      acc = wmma_f16(a, bm, acc);
    }
    half_t* base = XS + (mt * 16 + hi * 8) * XS_STRIDE + head * DHEAD + ntl * 16 + nl;
#pragma unroll
    for (int r = 0; r < 8; ++r) base[r * XS_STRIDE] = (half_t)acc[r];
  }
  __syncthreads();

  // ---- Phase 5: proj GEMM + bias, reverse-shift scatter to global ----
  {
    v16h a[4];
#pragma unroll
    for (int ks = 0; ks < 4; ++ks) a[ks] = frag_A(XS, XS_STRIDE, mt * 16, ks * 32);  // hoisted per wave
    for (int j = 0; j < 4; ++j) {
      int nt = nh * 4 + j;             // 0..7
      v8f acc = {0, 0, 0, 0, 0, 0, 0, 0};
#pragma unroll
      for (int ks = 0; ks < 4; ++ks) {
        v16h bm = frag_B(projw_h, CDIM, nt * 16, ks * 32);
        acc = wmma_f16(a[ks], bm, acc);
      }
      float bias = proj_bias[nt * 16 + nl];
      int m0 = mt * 16 + hi * 8;
#pragma unroll
      for (int r = 0; r < 8; ++r) {
        int M = m0 + r;
        if (M < 49) {
          int oy = (wy * WIN + M / 7 + SHIFTV) % HW;   // roll(out, +s)
          int ox = (wx * WIN + M % 7 + SHIFTV) % HW;
          out[(((size_t)b * HW + oy) * HW + ox) * CDIM + nt * 16 + nl] = acc[r] + bias;
        }
      }
    }
  }
}

extern "C" void kernel_launch(void* const* d_in, const int* in_sizes, int n_in,
                              void* d_out, int out_size, void* d_ws, size_t ws_size,
                              hipStream_t stream) {
  const float* x      = (const float*)d_in[0];
  const float* qkv_w  = (const float*)d_in[1];
  const float* qkv_b  = (const float*)d_in[2];
  const float* proj_w = (const float*)d_in[3];
  const float* proj_b = (const float*)d_in[4];
  const float* rel    = (const float*)d_in[5];

  half_t* qkvw_h  = (half_t*)d_ws;                  // 384*128 halves
  half_t* projw_h = qkvw_h + 3 * CDIM * CDIM;       // 128*128 halves

  swin_convert_w<<<(3 * CDIM * CDIM + 255) / 256, 256, 0, stream>>>(qkv_w, proj_w, qkvw_h, projw_h);

  size_t smem = (size_t)(3 * LPAD * XS_STRIDE + CDIM * VT_STRIDE + HEADS * LPAD * AT_STRIDE) * sizeof(half_t);
  (void)hipFuncSetAttribute((const void*)swin_window_attn,
                            hipFuncAttributeMaxDynamicSharedMemorySize, (int)smem);
  swin_window_attn<<<64 * 64, 256, smem, stream>>>(x, qkv_b, proj_b, rel, qkvw_h, projw_h, (float*)d_out);
}
